// Attention_3770981286778
// MI455X (gfx1250) — matrix operand order repaired and checked by reading, not compiled, not run on hardware
//
#include <hip/hip_runtime.h>
#include <hip/hip_bf16.h>
#include <stdint.h>

typedef __attribute__((ext_vector_type(16))) __bf16 v16bf;
typedef __attribute__((ext_vector_type(8)))  __bf16 v8bf;
typedef __attribute__((ext_vector_type(8)))  float  v8f;

#define DEV_INLINE __device__ __forceinline__

DEV_INLINE __bf16 f2bf(float f) {
  union { float f; uint32_t u; } in; in.f = f;
  uint32_t r = in.u + 0x7FFFu + ((in.u >> 16) & 1u);
  union { uint16_t u; __bf16 b; } out; out.u = (uint16_t)(r >> 16);
  return out.b;
}
DEV_INLINE float bf2f(__bf16 b) {
  union { uint16_t u; __bf16 b; } in; in.b = b;
  union { uint32_t u; float f; } out; out.u = ((uint32_t)in.u) << 16;
  return out.f;
}

// A-matrix fragment (16x32 bf16): lane half hf holds K chunks {hf*8..+7, 16+hf*8..+7}
DEV_INLINE v16bf frag_ld_a(const __bf16* base, int row, int stride, int k0, int hf) {
  const __bf16* p = base + row * stride + k0 + hf * 8;
  v8bf lo = *(const v8bf*)p;
  v8bf hi = *(const v8bf*)(p + 16);
  v16bf r;
#pragma unroll
  for (int i = 0; i < 8; ++i) { r[i] = lo[i]; r[i + 8] = hi[i]; }
  return r;
}
// B-matrix fragment (32x16 bf16): lane n=lane&15, half hf holds K chunks {hf*8..+7, 16+hf*8..+7}
DEV_INLINE v16bf frag_ld_b(const __bf16* base, int row, int stride, int k0, int hf) {
  const __bf16* p = base + row * stride + k0 + hf * 8;
  v8bf lo = *(const v8bf*)p;
  v8bf hi = *(const v8bf*)(p + 16);
  v16bf r;
#pragma unroll
  for (int i = 0; i < 8; ++i) { r[i] = lo[i]; r[i + 8] = hi[i]; }
  return r;
}
DEV_INLINE v8f wmma_bf16(v16bf a, v16bf b, v8f c) {
  return __builtin_amdgcn_wmma_f32_16x16x32_bf16(false, a, false, b, (short)0, c, false, false);
}
DEV_INLINE v8bf pack8(float4 a, float4 b) {
  v8bf r;
  r[0] = f2bf(a.x); r[1] = f2bf(a.y); r[2] = f2bf(a.z); r[3] = f2bf(a.w);
  r[4] = f2bf(b.x); r[5] = f2bf(b.y); r[6] = f2bf(b.z); r[7] = f2bf(b.w);
  return r;
}

// ---------------------------------------------------------------------------
// Kernel 1: QKV projection  Y = X @ Wqkv^T, scatter into Q(scaled)/K/V bf16
// X: (8192, 768) f32, W: (2304, 768) f32.  Double-buffered LDS k-loop.
// ---------------------------------------------------------------------------
__global__ __launch_bounds__(256)
void qkv_gemm(const float* __restrict__ X, const float* __restrict__ W,
              __bf16* __restrict__ Qp, __bf16* __restrict__ Kp,
              __bf16* __restrict__ Vp) {
  __shared__ __align__(16) __bf16 As[2][128][40];
  __shared__ __align__(16) __bf16 Bs[2][128][40];
  const int t = threadIdx.x;
  const int bm = blockIdx.x * 128;
  const int bn = blockIdx.y * 128;
  const int wid = t >> 5, lane = t & 31;
  const int wm = wid >> 2, wn = wid & 3;
  const int hf = lane >> 4, lrow = lane & 15;
  const int ldr = t >> 1, ldh = t & 1;

  const float* aSrc = X + (size_t)(bm + ldr) * 768 + ldh * 16;
  const float* bSrc = W + (size_t)(bn + ldr) * 768 + ldh * 16;

  v8f acc[4][2] = {};

  {  // preload tile 0
    const float4* a4 = (const float4*)aSrc;
    *(v8bf*)&As[0][ldr][ldh * 16]     = pack8(a4[0], a4[1]);
    *(v8bf*)&As[0][ldr][ldh * 16 + 8] = pack8(a4[2], a4[3]);
    const float4* b4 = (const float4*)bSrc;
    *(v8bf*)&Bs[0][ldr][ldh * 16]     = pack8(b4[0], b4[1]);
    *(v8bf*)&Bs[0][ldr][ldh * 16 + 8] = pack8(b4[2], b4[3]);
    __builtin_prefetch(aSrc + 32, 0, 0);
    __builtin_prefetch(bSrc + 32, 0, 0);
  }
  __syncthreads();

  for (int kt = 0; kt < 24; ++kt) {
    const int cur = kt & 1, nxt = cur ^ 1;
    const bool hasNext = (kt + 1) < 24;
    float4 f0{}, f1{}, f2{}, f3{}, g0{}, g1{}, g2{}, g3{};
    if (hasNext) {  // issue next tile's global loads before the matrix block
      const float4* a4 = (const float4*)(aSrc + (kt + 1) * 32);
      f0 = a4[0]; f1 = a4[1]; f2 = a4[2]; f3 = a4[3];
      const float4* b4 = (const float4*)(bSrc + (kt + 1) * 32);
      g0 = b4[0]; g1 = b4[1]; g2 = b4[2]; g3 = b4[3];
      if (kt + 2 < 24) {
        __builtin_prefetch(aSrc + (kt + 2) * 32, 0, 0);
        __builtin_prefetch(bSrc + (kt + 2) * 32, 0, 0);
      }
    }

    v16bf af[4], bfr[2];
#pragma unroll
    for (int i = 0; i < 4; ++i)
      af[i] = frag_ld_a(&As[cur][0][0], wm * 64 + 16 * i + lrow, 40, 0, hf);
#pragma unroll
    for (int j = 0; j < 2; ++j)
      bfr[j] = frag_ld_b(&Bs[cur][0][0], wn * 32 + 16 * j + lrow, 40, 0, hf);
#pragma unroll
    for (int i = 0; i < 4; ++i) {
#pragma unroll
      for (int j = 0; j < 2; ++j) acc[i][j] = wmma_bf16(af[i], bfr[j], acc[i][j]);
    }

    if (hasNext) {
      *(v8bf*)&As[nxt][ldr][ldh * 16]     = pack8(f0, f1);
      *(v8bf*)&As[nxt][ldr][ldh * 16 + 8] = pack8(f2, f3);
      *(v8bf*)&Bs[nxt][ldr][ldh * 16]     = pack8(g0, g1);
      *(v8bf*)&Bs[nxt][ldr][ldh * 16 + 8] = pack8(g2, g3);
    }
    __syncthreads();
  }

  // scatter: n -> (sel, h, d); m -> (b, nn). Q pre-scaled by 64^-0.5 = 0.125
#pragma unroll
  for (int i = 0; i < 4; ++i) {
#pragma unroll
    for (int j = 0; j < 2; ++j) {
#pragma unroll
      for (int r = 0; r < 8; ++r) {
        int m = bm + wm * 64 + 16 * i + r + 8 * hf;
        int n = bn + wn * 32 + 16 * j + lrow;
        int b = m >> 10, nn = m & 1023;
        int sel = n / 768, c = n % 768;
        int h = c >> 6, d = c & 63;
        float v = acc[i][j][r];
        size_t idx = ((((size_t)b * 12 + h) << 10) + nn) * 64 + d;
        if (sel == 0)      Qp[idx] = f2bf(v * 0.125f);
        else if (sel == 1) Kp[idx] = f2bf(v);
        else               Vp[idx] = f2bf(v);
      }
    }
  }
}

// ---------------------------------------------------------------------------
// Kernel 2: gate mask = sigmoid(mean_h(q_last . k))   (q already scaled)
// ---------------------------------------------------------------------------
__global__ __launch_bounds__(256)
void mask_kernel(const __bf16* __restrict__ Qp, const __bf16* __restrict__ Kp,
                 float* __restrict__ mask_out) {
  __shared__ float qlds[768];
  const int t = threadIdx.x;
  const int b = blockIdx.x >> 2;
  const int n = ((blockIdx.x & 3) << 8) + t;
  for (int i = t; i < 768; i += 256) {
    int h = i >> 6, d = i & 63;
    qlds[i] = bf2f(Qp[((((size_t)b * 12 + h) << 10) + 1023) * 64 + d]);
  }
  __syncthreads();
  float sum = 0.f;
  for (int h = 0; h < 12; ++h) {
    const v8bf* kr = (const v8bf*)(Kp + ((((size_t)b * 12 + h) << 10) + n) * 64);
    float a = 0.f;
#pragma unroll
    for (int c = 0; c < 8; ++c) {
      v8bf kv = kr[c];
#pragma unroll
      for (int i = 0; i < 8; ++i) a += qlds[h * 64 + c * 8 + i] * bf2f(kv[i]);
    }
    sum += a;
  }
  sum *= (1.f / 12.f);
  mask_out[(size_t)b * 1024 + n] = 1.f / (1.f + __expf(-sum));
}

// ---------------------------------------------------------------------------
// Kernel 3: flash attention with post-softmax sigmoid gating.
// grid = (N/128, B*H). 8 waves, each owns a 16-query strip.
// ---------------------------------------------------------------------------
__global__ __launch_bounds__(256)
void flash_attn(const __bf16* __restrict__ Qp, const __bf16* __restrict__ Kp,
                const __bf16* __restrict__ Vp, const float* __restrict__ gate,
                const int* __restrict__ cur_depth, __bf16* __restrict__ att) {
  __shared__ __align__(16) __bf16 Qs[128][72];
  __shared__ __align__(16) __bf16 Ks[128][72];
  __shared__ __align__(16) __bf16 Vt[64][136];
  __shared__ __align__(16) __bf16 Ps[128][136];
  __shared__ float gs[128];

  const int t = threadIdx.x;
  const int bh = blockIdx.y;
  const int b = bh / 12, h = bh % 12;
  const int q0 = blockIdx.x * 128;
  const bool gated = (cur_depth[0] >= 9);
  const int wid = t >> 5, lane = t & 31;
  const int hf = lane >> 4, lrow = lane & 15;
  const int ldr = t >> 1, ldh = t & 1;
  const size_t base = ((size_t)bh) << 10;  // row base in (B*H, N)

  {  // load Q tile (128 x 64)
    const v8bf* src = (const v8bf*)(Qp + (base + q0 + ldr) * 64 + ldh * 32);
    v8bf a = src[0], c = src[1], d = src[2], e = src[3];
    *(v8bf*)&Qs[ldr][ldh * 32]      = a;
    *(v8bf*)&Qs[ldr][ldh * 32 + 8]  = c;
    *(v8bf*)&Qs[ldr][ldh * 32 + 16] = d;
    *(v8bf*)&Qs[ldr][ldh * 32 + 24] = e;
    __builtin_prefetch(Kp + (base + ldr) * 64 + ldh * 32, 0, 0);
    __builtin_prefetch(Vp + (base + ldr) * 64 + ldh * 32, 0, 0);
  }

  v8f acc_o[4] = {};
  float mrun[8], lrun[8];
#pragma unroll
  for (int r = 0; r < 8; ++r) { mrun[r] = -1e30f; lrun[r] = 0.f; }

  for (int kt = 0; kt < 8; ++kt) {
    __syncthreads();
    {  // K tile (128 keys x 64) row-major
      const v8bf* src = (const v8bf*)(Kp + (base + kt * 128 + ldr) * 64 + ldh * 32);
      v8bf a = src[0], c = src[1], d = src[2], e = src[3];
      *(v8bf*)&Ks[ldr][ldh * 32]      = a;
      *(v8bf*)&Ks[ldr][ldh * 32 + 8]  = c;
      *(v8bf*)&Ks[ldr][ldh * 32 + 16] = d;
      *(v8bf*)&Ks[ldr][ldh * 32 + 24] = e;
    }
    {  // V tile transposed: Vt[d][key]
      const v8bf* src = (const v8bf*)(Vp + (base + kt * 128 + ldr) * 64 + ldh * 32);
#pragma unroll
      for (int c = 0; c < 4; ++c) {
        v8bf vv = src[c];
#pragma unroll
        for (int i = 0; i < 8; ++i) Vt[ldh * 32 + c * 8 + i][ldr] = vv[i];
      }
    }
    if (t < 128) gs[t] = gated ? gate[(size_t)b * 1024 + kt * 128 + t] : 1.f;
    if (kt + 1 < 8) {  // pull next tile toward GL2/L0 while we compute
      __builtin_prefetch(Kp + (base + (kt + 1) * 128 + ldr) * 64 + ldh * 32, 0, 0);
      __builtin_prefetch(Vp + (base + (kt + 1) * 128 + ldr) * 64 + ldh * 32, 0, 0);
    }
    __syncthreads();

    // S = Q K^T  (16 x 128 per wave)
    v8f s[8] = {};
#pragma unroll
    for (int kk = 0; kk < 2; ++kk) {
      v16bf aq = frag_ld_a(&Qs[0][0], wid * 16 + lrow, 72, kk * 32, hf);
#pragma unroll
      for (int j = 0; j < 8; ++j) {
        v16bf bk = frag_ld_b(&Ks[0][0], j * 16 + lrow, 72, kk * 32, hf);
        s[j] = wmma_bf16(aq, bk, s[j]);
      }
    }

    // online softmax; row r lives in 16-lane half (row = r + 8*hf)
#pragma unroll
    for (int r = 0; r < 8; ++r) {
      float tm = -1e30f;
#pragma unroll
      for (int j = 0; j < 8; ++j) tm = fmaxf(tm, s[j][r]);
#pragma unroll
      for (int off = 1; off < 16; off <<= 1) tm = fmaxf(tm, __shfl_xor(tm, off, 32));
      float mnew = fmaxf(mrun[r], tm);
      float alpha = __expf(mrun[r] - mnew);
      mrun[r] = mnew;
      lrun[r] *= alpha;
#pragma unroll
      for (int j = 0; j < 4; ++j) acc_o[j][r] *= alpha;
      float lp = 0.f;
#pragma unroll
      for (int j = 0; j < 8; ++j) {
        float p = __expf(s[j][r] - mnew);
        lp += p;  // denominator is UNgated (softmax first, then gate)
        Ps[wid * 16 + r + 8 * hf][j * 16 + lrow] = f2bf(p * gs[j * 16 + lrow]);
      }
#pragma unroll
      for (int off = 1; off < 16; off <<= 1) lp += __shfl_xor(lp, off, 32);
      lrun[r] += lp;
    }

    // O += P_gated @ V   (P strip is wave-private; LDS ops are in-order per wave)
#pragma unroll
    for (int kk = 0; kk < 4; ++kk) {
      v16bf ap = frag_ld_a(&Ps[0][0], wid * 16 + lrow, 136, kk * 32, hf);
#pragma unroll
      for (int j = 0; j < 4; ++j) {
        v16bf bv = frag_ld_b(&Vt[0][0], j * 16 + lrow, 136, kk * 32, hf);
        acc_o[j] = wmma_bf16(ap, bv, acc_o[j]);
      }
    }
  }

  // epilogue: att layout (B, N, H*hd) bf16 for the proj GEMM
#pragma unroll
  for (int j = 0; j < 4; ++j) {
#pragma unroll
    for (int r = 0; r < 8; ++r) {
      int m = wid * 16 + r + 8 * hf;
      int d = j * 16 + lrow;
      float v = acc_o[j][r] / lrun[r];
      size_t idx = ((((size_t)b << 10) + q0 + m)) * 768 + h * 64 + d;
      att[idx] = f2bf(v);
    }
  }
}

// ---------------------------------------------------------------------------
// Kernel 4: output projection  out = att @ proj_w^T + b  (double-buffered)
// ---------------------------------------------------------------------------
__global__ __launch_bounds__(256)
void proj_gemm(const __bf16* __restrict__ A, const float* __restrict__ W,
               const float* __restrict__ bias, float* __restrict__ out) {
  __shared__ __align__(16) __bf16 As[2][128][40];
  __shared__ __align__(16) __bf16 Bs[2][128][40];
  const int t = threadIdx.x;
  const int bm = blockIdx.x * 128;
  const int bn = blockIdx.y * 128;
  const int wid = t >> 5, lane = t & 31;
  const int wm = wid >> 2, wn = wid & 3;
  const int hf = lane >> 4, lrow = lane & 15;
  const int ldr = t >> 1, ldh = t & 1;

  const __bf16* aSrc = A + (size_t)(bm + ldr) * 768 + ldh * 16;
  const float*  bSrc = W + (size_t)(bn + ldr) * 768 + ldh * 16;

  v8f acc[4][2] = {};

  {  // preload tile 0
    const v8bf* a8 = (const v8bf*)aSrc;
    *(v8bf*)&As[0][ldr][ldh * 16]     = a8[0];
    *(v8bf*)&As[0][ldr][ldh * 16 + 8] = a8[1];
    const float4* b4 = (const float4*)bSrc;
    *(v8bf*)&Bs[0][ldr][ldh * 16]     = pack8(b4[0], b4[1]);
    *(v8bf*)&Bs[0][ldr][ldh * 16 + 8] = pack8(b4[2], b4[3]);
    __builtin_prefetch(aSrc + 32, 0, 0);
    __builtin_prefetch(bSrc + 32, 0, 0);
  }
  __syncthreads();

  for (int kt = 0; kt < 24; ++kt) {
    const int cur = kt & 1, nxt = cur ^ 1;
    const bool hasNext = (kt + 1) < 24;
    v8bf x0{}, x1{};
    float4 g0{}, g1{}, g2{}, g3{};
    if (hasNext) {
      const v8bf* a8 = (const v8bf*)(aSrc + (kt + 1) * 32);
      x0 = a8[0]; x1 = a8[1];
      const float4* b4 = (const float4*)(bSrc + (kt + 1) * 32);
      g0 = b4[0]; g1 = b4[1]; g2 = b4[2]; g3 = b4[3];
      if (kt + 2 < 24) {
        __builtin_prefetch(aSrc + (kt + 2) * 32, 0, 0);
        __builtin_prefetch(bSrc + (kt + 2) * 32, 0, 0);
      }
    }

    v16bf af[4], bfr[2];
#pragma unroll
    for (int i = 0; i < 4; ++i)
      af[i] = frag_ld_a(&As[cur][0][0], wm * 64 + 16 * i + lrow, 40, 0, hf);
#pragma unroll
    for (int j = 0; j < 2; ++j)
      bfr[j] = frag_ld_b(&Bs[cur][0][0], wn * 32 + 16 * j + lrow, 40, 0, hf);
#pragma unroll
    for (int i = 0; i < 4; ++i) {
#pragma unroll
      for (int j = 0; j < 2; ++j) acc[i][j] = wmma_bf16(af[i], bfr[j], acc[i][j]);
    }

    if (hasNext) {
      *(v8bf*)&As[nxt][ldr][ldh * 16]     = x0;
      *(v8bf*)&As[nxt][ldr][ldh * 16 + 8] = x1;
      *(v8bf*)&Bs[nxt][ldr][ldh * 16]     = pack8(g0, g1);
      *(v8bf*)&Bs[nxt][ldr][ldh * 16 + 8] = pack8(g2, g3);
    }
    __syncthreads();
  }

#pragma unroll
  for (int i = 0; i < 4; ++i) {
#pragma unroll
    for (int j = 0; j < 2; ++j) {
#pragma unroll
      for (int r = 0; r < 8; ++r) {
        int m = bm + wm * 64 + 16 * i + r + 8 * hf;
        int n = bn + wn * 32 + 16 * j + lrow;
        out[(size_t)m * 768 + n] = acc[i][j][r] + bias[n];
      }
    }
  }
}

// ---------------------------------------------------------------------------
extern "C" void kernel_launch(void* const* d_in, const int* in_sizes, int n_in,
                              void* d_out, int out_size, void* d_ws, size_t ws_size,
                              hipStream_t stream) {
  (void)in_sizes; (void)n_in; (void)out_size; (void)ws_size;
  const float* x      = (const float*)d_in[0];
  const float* qkv_w  = (const float*)d_in[1];
  const float* proj_w = (const float*)d_in[2];
  const float* proj_b = (const float*)d_in[3];
  const int*   depth  = (const int*)d_in[4];

  float* out  = (float*)d_out;                   // (8,1024,768) f32
  float* mask = out + (size_t)8 * 1024 * 768;    // (8,1024)    f32

  const size_t qkv_elems = (size_t)8 * 12 * 1024 * 64;  // per Q/K/V
  __bf16* Qp  = (__bf16*)d_ws;
  __bf16* Kp  = Qp + qkv_elems;
  __bf16* Vp  = Kp + qkv_elems;
  __bf16* att = Vp + qkv_elems;  // (8192, 768) bf16

  qkv_gemm <<<dim3(64, 18), 256, 0, stream>>>(x, qkv_w, Qp, Kp, Vp);
  mask_kernel<<<dim3(32),    256, 0, stream>>>(Qp, Kp, mask);
  flash_attn <<<dim3(8, 96), 256, 0, stream>>>(Qp, Kp, Vp, mask, depth, att);
  proj_gemm  <<<dim3(64, 6), 256, 0, stream>>>(att, proj_w, proj_b, out);
}
